// SelfAttentionBlock_64304250356464
// MI455X (gfx1250) — matrix operand order repaired and checked by reading, not compiled, not run on hardware
//
#include <hip/hip_runtime.h>

// ---------------------------------------------------------------------------
// CDNA5 (gfx1250) wave32 WMMA implementation of a SAGAN-style non-local block.
// All GEMMs use v_wmma_f32_16x16x32_bf16 (f32 accumulate). Attention is
// streamed flash-style. Weights / pooled operands are pre-packed into WMMA
// fragment order. Fragment batches are consumed in reverse load order so all
// loads of a batch issue as one clause with a single wait (no double-buffer
// register blowup / spills).
// ---------------------------------------------------------------------------

typedef __attribute__((ext_vector_type(16))) __bf16 bf16x16;
typedef __attribute__((ext_vector_type(8)))  __bf16 bf16x8;
typedef __attribute__((ext_vector_type(8)))  float  v8f;
typedef __attribute__((ext_vector_type(4)))  float  f32x4;

#define DEV __device__ __forceinline__

DEV __bf16 f2bf(float f) {
  unsigned u = __builtin_bit_cast(unsigned, f);
  unsigned r = u + 0x7FFFu + ((u >> 16) & 1u);   // round-to-nearest-even
  unsigned short h = (unsigned short)(r >> 16);
  return __builtin_bit_cast(__bf16, h);
}
DEV float bf2f(__bf16 b) {
  unsigned short h = __builtin_bit_cast(unsigned short, b);
  unsigned u = ((unsigned)h) << 16;
  return __builtin_bit_cast(float, u);
}

DEV v8f wmma_bf16(bf16x16 a, bf16x16 b, v8f c) {
  return __builtin_amdgcn_wmma_f32_16x16x32_bf16(false, a, false, b, (short)0, c,
                                                 false, false);
}

// ---- Fragment loaders ------------------------------------------------------
// Packed fragment: 32 lanes x 16 bf16, lane's data contiguous (32B -> 2xb128).
DEV bf16x16 loadFrag(const __bf16* fragBase, int lane) {
  return *(const bf16x16*)(fragBase + (size_t)lane * 16);
}
// A fragment 16x32 (MxK) from row-major bf16, ld = row stride.
// lane L: row = L&15, half = L>>4; elements e<8 -> K=e+8*half; e>=8 -> K=e+8+8*half
DEV bf16x16 loadA_bf(const __bf16* p, int ld, int lane) {
  int m = lane & 15, half = lane >> 4;
  const __bf16* row = p + (size_t)m * ld + 8 * half;
  bf16x8 lo = *(const bf16x8*)(row);
  bf16x8 hi = *(const bf16x8*)(row + 16);
  return __builtin_shufflevector(lo, hi, 0, 1, 2, 3, 4, 5, 6, 7,
                                 8, 9, 10, 11, 12, 13, 14, 15);
}
// A fragment from row-major f32 (vectorized loads, then batched converts).
DEV bf16x16 loadA_f32(const float* p, int ld, int lane) {
  int m = lane & 15, half = lane >> 4;
  const float* row = p + (size_t)m * ld + 8 * half;
  f32x4 q0 = *(const f32x4*)(row);
  f32x4 q1 = *(const f32x4*)(row + 4);
  f32x4 q2 = *(const f32x4*)(row + 16);
  f32x4 q3 = *(const f32x4*)(row + 20);
  bf16x16 a;
#pragma unroll
  for (int e = 0; e < 4; ++e) {
    a[e]      = f2bf(q0[e]);
    a[e + 4]  = f2bf(q1[e]);
    a[e + 8]  = f2bf(q2[e]);
    a[e + 12] = f2bf(q3[e]);
  }
  return a;
}
// B fragment where source stores B-transposed row-major [N x K]:
// lane L: col = L&15, half = L>>4; e<8 -> K = e + 8*half; e>=8 -> K = e+8 + 8*half.
DEV bf16x16 loadBt_bf(const __bf16* p, int ld, int lane) {
  int n = lane & 15, half = lane >> 4;
  const __bf16* row = p + (size_t)n * ld + 8 * half;
  bf16x8 lo = *(const bf16x8*)(row);
  bf16x8 hi = *(const bf16x8*)(row + 16);
  return __builtin_shufflevector(lo, hi, 0, 1, 2, 3, 4, 5, 6, 7,
                                 8, 9, 10, 11, 12, 13, 14, 15);
}

// ---------------------------------------------------------------------------
// W0: pack a f32 weight [K x N] into bf16 B-fragment order.
// Fragment f = kc*(N/16) + nt: element (lane, e):
//   k = kc*32 + (e<8 ? e : e+8) + 8*(lane>>4), n = nt*16 + (lane&15)
// ---------------------------------------------------------------------------
__global__ void pack_w_kernel(const float* __restrict__ w, __bf16* __restrict__ out,
                              int K, int N) {
  int t = blockIdx.x * blockDim.x + threadIdx.x;
  int lane = t & 31;
  int f = t >> 5;
  int NT = N >> 4;
  if (f >= (K >> 5) * NT) return;
  int kc = f / NT, nt = f % NT;
  int n = nt * 16 + (lane & 15);
  int kbase = kc * 32 + 8 * (lane >> 4);
  __bf16* o = out + ((size_t)f * 32 + lane) * 16;
#pragma unroll
  for (int e = 0; e < 16; ++e)
    o[e] = f2bf(w[(size_t)(kbase + (e < 8 ? e : e + 8)) * N + n]);
}

// ---------------------------------------------------------------------------
// K1: fused projection GEMM. X[4096x512] @ {theta|phi|g} + bias -> bf16.
// Per wave: 16 rows x 64 cols. Batched B loads (consumed in reverse order,
// single wait) + double-buffered A.
// ---------------------------------------------------------------------------
__global__ void proj_kernel(const float* __restrict__ x,
                            const __bf16* __restrict__ wpTheta,
                            const __bf16* __restrict__ wpPhi,
                            const __bf16* __restrict__ wpG,
                            const float* __restrict__ b_theta,
                            const float* __restrict__ b_phi,
                            const float* __restrict__ b_g,
                            __bf16* __restrict__ theta, __bf16* __restrict__ phiF,
                            __bf16* __restrict__ gF) {
  int wave = blockIdx.x * (blockDim.x >> 5) + (threadIdx.x >> 5);
  int lane = threadIdx.x & 31;
  int grp = wave % 6;
  int rt  = (wave / 6) & 255;
  int b   = wave / (6 * 256);

  const __bf16* wp; const float* bias; int NT, ntbase, colbase, ldo; __bf16* outp;
  if (grp == 0) {
    wp = wpTheta; bias = b_theta; NT = 4; ntbase = 0; colbase = 0; ldo = 64;
    outp = theta + ((size_t)b * 4096 + rt * 16) * 64;
  } else if (grp == 1) {
    wp = wpPhi; bias = b_phi; NT = 4; ntbase = 0; colbase = 0; ldo = 64;
    outp = phiF + ((size_t)b * 4096 + rt * 16) * 64;
  } else {
    wp = wpG; bias = b_g; NT = 16; ntbase = (grp - 2) * 4; colbase = (grp - 2) * 64;
    ldo = 256;
    outp = gF + ((size_t)b * 4096 + rt * 16) * 256 + colbase;
  }

  const float* xp = x + ((size_t)b * 4096 + rt * 16) * 512;
  v8f acc[4] = {};
  bf16x16 a = loadA_f32(xp, 512, lane);
  for (int kc = 0; kc < 16; ++kc) {
    bf16x16 bb[4];
#pragma unroll
    for (int t = 0; t < 4; ++t)
      bb[t] = loadFrag(wp + (size_t)(kc * NT + ntbase + t) * 512, lane);
    bf16x16 an = a;
    if (kc < 15) an = loadA_f32(xp + (kc + 1) * 32, 512, lane);  // prefetch next A
    // reverse order: bb[3] (loaded last) first => one wait covers the batch
#pragma unroll
    for (int t = 3; t >= 0; --t) acc[t] = wmma_bf16(a, bb[t], acc[t]);
    a = an;
  }
  int n = lane & 15, half = lane >> 4;
#pragma unroll
  for (int t = 0; t < 4; ++t) {
    float bv = bias[colbase + t * 16 + n];
#pragma unroll
    for (int r = 0; r < 8; ++r) {
      int m = r + 8 * half;  // C/D layout: VGPR r -> row r + 8*half, col = lane&15
      outp[(size_t)m * ldo + t * 16 + n] = f2bf(acc[t][r] + bv);
    }
  }
}

// ---------------------------------------------------------------------------
// K2a: 2x2 maxpool phi [B,64,64,64] -> A-fragment-packed [b][kt(64)][c(2)][lane][16]
// ---------------------------------------------------------------------------
__global__ void pool_pack_phi(const __bf16* __restrict__ phiF,
                              __bf16* __restrict__ phiPp) {
  int t = blockIdx.x * blockDim.x + threadIdx.x;  // 524288
  int e    = t & 15;
  int lane = (t >> 4) & 31;
  int c    = (t >> 9) & 1;
  int kt   = (t >> 10) & 63;
  int b    = t >> 16;
  int d = c * 32 + (e < 8 ? e : e + 8) + 8 * (lane >> 4);
  int key = kt * 16 + (lane & 15);
  int hp = key >> 5, wp = key & 31;
  size_t base = (((size_t)b * 4096) + (size_t)(hp * 2) * 64 + wp * 2) * 64 + d;
  float m = fmaxf(fmaxf(bf2f(phiF[base]), bf2f(phiF[base + 64])),
                  fmaxf(bf2f(phiF[base + (size_t)64 * 64]),
                        bf2f(phiF[base + (size_t)64 * 64 + 64])));
  phiPp[t] = f2bf(m);
}

// ---------------------------------------------------------------------------
// K2b: 2x2 maxpool g [B,64,64,256] -> B-fragment-packed [b][kc(32)][dt(16)][lane][16]
// ---------------------------------------------------------------------------
__global__ void pool_pack_g(const __bf16* __restrict__ gF,
                            __bf16* __restrict__ gPp) {
  int t = blockIdx.x * blockDim.x + threadIdx.x;  // 2097152
  int e    = t & 15;
  int lane = (t >> 4) & 31;
  int dt   = (t >> 9) & 15;
  int kc   = (t >> 13) & 31;
  int b    = t >> 18;
  int key = kc * 32 + (e < 8 ? e : e + 8) + 8 * (lane >> 4);
  int d = dt * 16 + (lane & 15);
  int hp = key >> 5, wp = key & 31;
  size_t base = (((size_t)b * 4096) + (size_t)(hp * 2) * 64 + wp * 2) * 256 + d;
  float m = fmaxf(fmaxf(bf2f(gF[base]), bf2f(gF[base + 256])),
                  fmaxf(bf2f(gF[base + (size_t)64 * 256]),
                        bf2f(gF[base + (size_t)64 * 256 + 256])));
  gPp[t] = f2bf(m);
}

// ---------------------------------------------------------------------------
// K3: flash attention. One wave per (batch, 16-query tile, d-half).
// S' = phi @ theta^T: C fragment has queries-in-lanes / keys-in-rows, so
// exp() of two consecutive 16-key tiles packed into v16bf elements 0..7/8..15
// IS the A fragment of the P@g WMMA. Each wave owns 128 of 256 g-columns
// (64 accumulator VGPRs). Next chunk's phi loads + a g prefetch issue under
// the exp block; g fragment groups are consumed in reverse load order.
// ---------------------------------------------------------------------------
__global__ void attn_kernel(const __bf16* __restrict__ theta,
                            const __bf16* __restrict__ phiPp,
                            const __bf16* __restrict__ gPp,
                            __bf16* __restrict__ attnG) {
  int wave = blockIdx.x * (blockDim.x >> 5) + (threadIdx.x >> 5);
  int lane = threadIdx.x & 31;
  int dh = wave & 1;          // d-half: columns dh*128 .. dh*128+127
  int qt = (wave >> 1) & 255;
  int b  = wave >> 9;

  const __bf16* thp = theta + ((size_t)b * 4096 + qt * 16) * 64;
  const __bf16* php = phiPp + (size_t)b * 64 * 1024;   // [kt][c][lane][16]
  const __bf16* gp  = gPp + (size_t)b * 512 * 512;     // [kc][dt][lane][16]

  // theta^T as B fragments (resident), K = 64 = 2 chunks
  bf16x16 tb0 = loadBt_bf(thp + 0,  64, lane);
  bf16x16 tb1 = loadBt_bf(thp + 32, 64, lane);

  // ---- pass 1: per-query max over all 1024 keys (double-buffered A) ----
  float mmax = -3.0e38f;
  {
    bf16x16 a0 = loadFrag(php, lane);
    bf16x16 a1 = loadFrag(php + 512, lane);
    for (int kt = 0; kt < 64; ++kt) {
      bf16x16 n0 = a0, n1 = a1;
      if (kt < 63) {
        const __bf16* fb = php + (size_t)(kt + 1) * 1024;
        n0 = loadFrag(fb, lane);
        n1 = loadFrag(fb + 512, lane);
      }
      v8f c = {};
      c = wmma_bf16(a0, tb0, c);
      c = wmma_bf16(a1, tb1, c);
#pragma unroll
      for (int r = 0; r < 8; ++r) mmax = fmaxf(mmax, c[r]);
      a0 = n0; a1 = n1;
    }
  }
  mmax = fmaxf(mmax, __shfl_xor(mmax, 16, 32));  // combine key halves per query

  // ---- pass 2: exp + accumulate P@g over 32-key chunks ----
  float lsum = 0.0f;
  v8f acc[8] = {};
  bf16x16 pa0 = loadFrag(php, lane);
  bf16x16 pa1 = loadFrag(php + 512, lane);
  bf16x16 pa2 = loadFrag(php + 1024, lane);
  bf16x16 pa3 = loadFrag(php + 1536, lane);
  for (int kc = 0; kc < 32; ++kc) {
    // logits for this chunk
    v8f clo = {}, chi = {};
    clo = wmma_bf16(pa0, tb0, clo);
    clo = wmma_bf16(pa1, tb1, clo);
    chi = wmma_bf16(pa2, tb0, chi);
    chi = wmma_bf16(pa3, tb1, chi);
    if (kc < 31) {  // next chunk's phi loads under the exp + g WMMAs
      const __bf16* fb = php + (size_t)(kc + 1) * 2048;
      pa0 = loadFrag(fb, lane);
      pa1 = loadFrag(fb + 512, lane);
      pa2 = loadFrag(fb + 1024, lane);
      pa3 = loadFrag(fb + 1536, lane);
      // prefetch next chunk's g fragments (8KB spread over the wave)
      const __bf16* gn = gp + (size_t)(kc + 1) * 8192 + (size_t)dh * 4096;
      __builtin_prefetch(gn + (size_t)lane * 128, 0, 0);
    }
    // exp + pack (runs while phi/g loads are in flight)
    bf16x16 p;
#pragma unroll
    for (int r = 0; r < 8; ++r) {
      float e0 = __expf(clo[r] - mmax);   // key kc*32 + r + 8*half
      float e1 = __expf(chi[r] - mmax);   // key kc*32 + 16 + r + 8*half
      lsum += e0 + e1;
      p[r]     = f2bf(e0);                // A-frag e<8  -> K = e + 8*half
      p[r + 8] = f2bf(e1);                // A-frag e>=8 -> K = e+8 + 8*half
    }
    const __bf16* gk = gp + (size_t)kc * 8192 + (size_t)dh * 4096;
    bf16x16 bb[4];
#pragma unroll
    for (int i = 0; i < 4; ++i) bb[i] = loadFrag(gk + (size_t)i * 512, lane);
#pragma unroll
    for (int i = 3; i >= 0; --i) acc[i] = wmma_bf16(p, bb[i], acc[i]);
#pragma unroll
    for (int i = 0; i < 4; ++i) bb[i] = loadFrag(gk + (size_t)(4 + i) * 512, lane);
#pragma unroll
    for (int i = 3; i >= 0; --i) acc[4 + i] = wmma_bf16(p, bb[i], acc[4 + i]);
  }
  lsum += __shfl_xor(lsum, 16, 32);  // full softmax denominator for query lane&15

  int n = lane & 15, half = lane >> 4;
  __bf16* outp = attnG + ((size_t)b * 4096 + qt * 16) * 256 + dh * 128;
#pragma unroll
  for (int r = 0; r < 8; ++r) {
    int m = r + 8 * half;                  // accumulator row = query m
    float inv = 1.0f / __shfl(lsum, m, 32);
#pragma unroll
    for (int dt = 0; dt < 8; ++dt) {
      outp[(size_t)m * 256 + dt * 16 + n] = f2bf(acc[dt][r] * inv);
    }
  }
}

// ---------------------------------------------------------------------------
// K4: out = x + sigma * (attn_g @ w_o + b_o). Per wave: 16 rows x 64 cols.
// Batched B loads (reverse consumption) + double-buffered A.
// ---------------------------------------------------------------------------
__global__ void outproj_kernel(const __bf16* __restrict__ attnG,
                               const __bf16* __restrict__ wpO,
                               const float* __restrict__ b_o, const float* __restrict__ x,
                               const float* __restrict__ sigma, float* __restrict__ out) {
  int wave = blockIdx.x * (blockDim.x >> 5) + (threadIdx.x >> 5);
  int lane = threadIdx.x & 31;
  int cg = wave & 7;
  int rt = (wave >> 3) & 255;
  int b  = wave >> 11;

  const __bf16* ap = attnG + ((size_t)b * 4096 + rt * 16) * 256;
  v8f acc[4] = {};
  bf16x16 a = loadA_bf(ap, 256, lane);
  for (int kc = 0; kc < 8; ++kc) {
    bf16x16 bb[4];
#pragma unroll
    for (int t = 0; t < 4; ++t)  // w_o packed: K=256 (8 kc), N=512 (32 nt)
      bb[t] = loadFrag(wpO + (size_t)(kc * 32 + cg * 4 + t) * 512, lane);
    bf16x16 an = a;
    if (kc < 7) an = loadA_bf(ap + (kc + 1) * 32, 256, lane);
#pragma unroll
    for (int t = 3; t >= 0; --t) acc[t] = wmma_bf16(a, bb[t], acc[t]);
    a = an;
  }
  float sg = sigma[0];
  int n = lane & 15, half = lane >> 4;
  size_t rowbase = (size_t)b * 4096 + rt * 16;
#pragma unroll
  for (int t = 0; t < 4; ++t) {
    int col = cg * 64 + t * 16 + n;
    float bv = b_o[col];
#pragma unroll
    for (int r = 0; r < 8; ++r) {
      int m = r + 8 * half;
      size_t idx = (rowbase + m) * 512 + col;
      out[idx] = x[idx] + sg * (acc[t][r] + bv);
    }
  }
}

// ---------------------------------------------------------------------------
extern "C" void kernel_launch(void* const* d_in, const int* in_sizes, int n_in,
                              void* d_out, int out_size, void* d_ws, size_t ws_size,
                              hipStream_t stream) {
  const float* x       = (const float*)d_in[0];
  const float* w_theta = (const float*)d_in[1];
  const float* b_theta = (const float*)d_in[2];
  const float* w_phi   = (const float*)d_in[3];
  const float* b_phi   = (const float*)d_in[4];
  const float* w_g     = (const float*)d_in[5];
  const float* b_g     = (const float*)d_in[6];
  const float* w_o     = (const float*)d_in[7];
  const float* b_o     = (const float*)d_in[8];
  const float* sigma   = (const float*)d_in[9];
  float* out = (float*)d_out;

  // Workspace layout (~29.6 MB). attnG aliases gF (gF dead after pooling).
  char* ws = (char*)d_ws;
  size_t off = 0;
  __bf16* theta = (__bf16*)(ws + off); off += (size_t)8 * 4096 * 64 * 2;   // 4 MB
  __bf16* phiF  = (__bf16*)(ws + off); off += (size_t)8 * 4096 * 64 * 2;   // 4 MB
  __bf16* gF    = (__bf16*)(ws + off); off += (size_t)8 * 4096 * 256 * 2;  // 16 MB
  __bf16* attnG = gF;  // alias: gF consumed by pooling before attnG written
  __bf16* phiPp = (__bf16*)(ws + off); off += (size_t)8 * 64 * 2 * 512 * 2;   // 1 MB
  __bf16* gPp   = (__bf16*)(ws + off); off += (size_t)8 * 512 * 512 * 2;      // 4 MB
  __bf16* wpTheta = (__bf16*)(ws + off); off += (size_t)64 * 512 * 2;   // 64 KB
  __bf16* wpPhi   = (__bf16*)(ws + off); off += (size_t)64 * 512 * 2;   // 64 KB
  __bf16* wpG     = (__bf16*)(ws + off); off += (size_t)256 * 512 * 2;  // 256 KB
  __bf16* wpO     = (__bf16*)(ws + off); off += (size_t)256 * 512 * 2;  // 256 KB

  // W0: pack weights into B-fragment order
  pack_w_kernel<<<(64 * 32 + 255) / 256, 256, 0, stream>>>(w_theta, wpTheta, 512, 64);
  pack_w_kernel<<<(64 * 32 + 255) / 256, 256, 0, stream>>>(w_phi, wpPhi, 512, 64);
  pack_w_kernel<<<(256 * 32 + 255) / 256, 256, 0, stream>>>(w_g, wpG, 512, 256);
  pack_w_kernel<<<(256 * 32 + 255) / 256, 256, 0, stream>>>(w_o, wpO, 256, 512);

  // K1: projections (8 * 256 row-tiles * 6 col-groups = 12288 waves)
  proj_kernel<<<12288 / 4, 128, 0, stream>>>(x, wpTheta, wpPhi, wpG,
                                             b_theta, b_phi, b_g,
                                             theta, phiF, gF);
  // K2: maxpool + fragment-pack phi and g
  pool_pack_phi<<<524288 / 256, 256, 0, stream>>>(phiF, phiPp);
  pool_pack_g<<<2097152 / 256, 256, 0, stream>>>(gF, gPp);

  // K3: flash attention (8 * 256 query tiles * 2 d-halves = 4096 waves)
  attn_kernel<<<4096 / 4, 128, 0, stream>>>(theta, phiPp, gPp, attnG);

  // K4: output projection + residual (8 * 256 * 8 = 16384 waves)
  outproj_kernel<<<16384 / 4, 128, 0, stream>>>(attnG, wpO, b_o, x, sigma, out);
}
